// EncoderGAT_3917010174724
// MI455X (gfx1250) — compile-verified, matrix operand order checked by reading, not run on hardware
//
#include <hip/hip_runtime.h>
#include <hip/hip_bf16.h>

// EncoderGAT for MI455X (gfx1250, wave32, WMMA), round 2.
// All layers run transposed: X^T_{l+1} = W^T_l x X^T_l, so the WMMA D registers
// of one layer ARE the B fragment of the next (lane = batch row, elements =
// feature runs matching the 16-bit B layout). Activations never touch LDS;
// weights are staged once per block in fragment-major LDS (2x ds_load_b128 per
// operand). Activations use branch-free v_exp_f32/v_rcp_f32 sequences.
//
// Workspace (f32 unless noted), requires (NA + 5*NS)*64*4 B ~ 77 MB:
//   uenc [NA,64] | msg [NS,64] | sum_u [NS,64] | mxEnc u32 [NS,64] | z [NS,64] | acc [NS,64]

#define HID 64

typedef __attribute__((ext_vector_type(16))) _Float16 v16h;
typedef __attribute__((ext_vector_type(8)))  _Float16 v8h;
typedef __attribute__((ext_vector_type(2)))  _Float16 v2h;
typedef __attribute__((ext_vector_type(8)))  float    v8f;

#define WMMA_F16(a, b, c) \
  __builtin_amdgcn_wmma_f32_16x16x32_f16(false, (a), false, (b), (short)0, (c), false, false)

// ---------------- fast, branch-free activations ----------------
#define LOG2E 1.44269504088896f

__device__ __forceinline__ float fast_exp(float x) {
  return __builtin_amdgcn_exp2f(x * LOG2E);
}
__device__ __forceinline__ float fast_tanh(float x) {
  float e = __builtin_amdgcn_exp2f(x * (2.0f * LOG2E));  // exp(2x); inf/0 at extremes -> +-1
  return 1.0f - 2.0f * __builtin_amdgcn_rcpf(e + 1.0f);
}
__device__ __forceinline__ float fast_sigmoid(float x) {
  return __builtin_amdgcn_rcpf(1.0f + __builtin_amdgcn_exp2f(-LOG2E * x));
}
__device__ __forceinline__ float apply_act(float v, int kind) {
  if (kind == 1) return fast_tanh(v);
  if (kind == 2) return fast_sigmoid(v);
  return v;
}

// ---------------- fragment helpers ----------------

// Load a pre-swizzled fragment: 16 contiguous halves per lane (32B, aligned).
__device__ __forceinline__ v16h frag_ld(const _Float16* p) {
  v8h lo = *(const v8h*)(p);
  v8h hi = *(const v8h*)(p + 8);
  return __builtin_shufflevector(lo, hi, 0, 1, 2, 3, 4, 5, 6, 7, 8, 9, 10, 11, 12, 13, 14, 15);
}

// B fragment from a row-major f16 row: elements 0..7 at p[0..7], 8..15 at p[16..23].
__device__ __forceinline__ v16h frag_ld_gap(const _Float16* p) {
  v8h lo = *(const v8h*)(p);
  v8h hi = *(const v8h*)(p + 16);
  return __builtin_shufflevector(lo, hi, 0, 1, 2, 3, 4, 5, 6, 7, 8, 9, 10, 11, 12, 13, 14, 15);
}

// B fragment from a row-major f32 global row (2x 32B vector loads + cvt).
__device__ __forceinline__ v16h frag_ld_rowf32(const float* rowp, int k0, int lane) {
  int kb = k0 + ((lane & 16) ? 8 : 0);
  float4 a0 = *(const float4*)(rowp + kb);
  float4 a1 = *(const float4*)(rowp + kb + 4);
  float4 c0 = *(const float4*)(rowp + kb + 16);
  float4 c1 = *(const float4*)(rowp + kb + 20);
  v16h b;
  b[0] = (_Float16)a0.x; b[1] = (_Float16)a0.y; b[2] = (_Float16)a0.z; b[3] = (_Float16)a0.w;
  b[4] = (_Float16)a1.x; b[5] = (_Float16)a1.y; b[6] = (_Float16)a1.z; b[7] = (_Float16)a1.w;
  b[8] = (_Float16)c0.x; b[9] = (_Float16)c0.y; b[10] = (_Float16)c0.z; b[11] = (_Float16)c0.w;
  b[12] = (_Float16)c1.x; b[13] = (_Float16)c1.y; b[14] = (_Float16)c1.z; b[15] = (_Float16)c1.w;
  return b;
}

__device__ __forceinline__ void zero4(v8f d[4]) {
#pragma unroll
  for (int n = 0; n < 4; ++n)
#pragma unroll
    for (int i = 0; i < 8; ++i) d[n][i] = 0.0f;
}

// Stage W^T (w row-major [kact][64] f32) into fragment-major f16 LDS:
// dst[((kc*4+m)*32+lane)*16+e] = w[f][o], o = m*16+(lane&15),
// f = kc*32 + 8*(lane>=16) + (e<8 ? e : 16+e-8); zero-pad f >= kact.
__device__ __forceinline__ void stage_wT(_Float16* dst, const float* w, int kact, int kch,
                                         int tid, int nth) {
  int total = kch * 4 * 32 * 16;
  for (int idx = tid; idx < total; idx += nth) {
    int e = idx & 15;
    int ln = (idx >> 4) & 31;
    int fi = idx >> 9;
    int kc = fi >> 2, m = fi & 3;
    int o = m * 16 + (ln & 15);
    int f = kc * 32 + ((ln & 16) ? 8 : 0) + ((e < 8) ? e : (16 + e - 8));
    dst[idx] = (f < kact) ? (_Float16)w[f * HID + o] : (_Float16)0.0f;
  }
}
__device__ __forceinline__ void stage_bias(float* dst, const float* src, int tid, int nth) {
  for (int i = tid; i < HID; i += nth) dst[i] = src[i];
}

// d[m][i] holds feature m*16 + rb + i of batch row (lane&15); bias add is two
// contiguous float4 LDS loads per m-tile.
__device__ __forceinline__ void add_bias4(v8f d[4], const float* sBias, int lane) {
  int rb = (lane & 16) ? 8 : 0;
#pragma unroll
  for (int m = 0; m < 4; ++m) {
    float4 b0 = *(const float4*)(sBias + m * 16 + rb);
    float4 b1 = *(const float4*)(sBias + m * 16 + rb + 4);
    d[m][0] += b0.x; d[m][1] += b0.y; d[m][2] += b0.z; d[m][3] += b0.w;
    d[m][4] += b1.x; d[m][5] += b1.y; d[m][6] += b1.z; d[m][7] += b1.w;
  }
}

// D-of-previous-layer -> B-of-next-layer, entirely in registers.
__device__ __forceinline__ v16h make_b(const v8f dlo, const v8f dhi, int kind) {
  v16h b;
#pragma unroll
  for (int e = 0; e < 8; ++e) {
    b[e]     = (_Float16)apply_act(dlo[e], kind);
    b[e + 8] = (_Float16)apply_act(dhi[e], kind);
  }
  return b;
}

// One K-chunk: dout[m] += W^T frag(m,kc) x b.
__device__ __forceinline__ void chunk_acc(v16h b, int kc, const _Float16* sWf, int lane,
                                          v8f dout[4]) {
#pragma unroll
  for (int m = 0; m < 4; ++m) {
    v16h aw = frag_ld(sWf + (((kc << 2) + m) * 32 + lane) * 16);
    dout[m] = WMMA_F16(aw, b, dout[m]);
  }
}

// Full 64->64 transposed layer; kind = activation of the *previous* layer.
__device__ __forceinline__ void layerT64(const v8f dprev[4], int kind, const _Float16* sWf,
                                         int lane, v8f dout[4]) {
  zero4(dout);
#pragma unroll
  for (int kc = 0; kc < 2; ++kc) {
    v16h b = make_b(dprev[2 * kc], dprev[2 * kc + 1], kind);
    chunk_acc(b, kc, sWf, lane, dout);
  }
}

// Order-preserving f32 <-> u32 for atomicMax (0 == -inf).
__device__ __forceinline__ unsigned fenc(float f) {
  unsigned u = __float_as_uint(f);
  return (u & 0x80000000u) ? ~u : (u | 0x80000000u);
}
__device__ __forceinline__ float fdec(unsigned u) {
  unsigned v = (u & 0x80000000u) ? (u & 0x7FFFFFFFu) : ~u;
  return __uint_as_float(v);
}

// ---------------- node: uenc = MLP_uu(u), [8->64->64->64], no bias ----------------
__global__ __launch_bounds__(256) void k_node_uu(const float* __restrict__ u,
                                                 const float* __restrict__ w0,
                                                 const float* __restrict__ w1,
                                                 const float* __restrict__ w2,
                                                 float* __restrict__ uenc, int na) {
  __shared__ __align__(16) _Float16 sW0[1 * 4 * 32 * 16];
  __shared__ __align__(16) _Float16 sW1[2 * 4 * 32 * 16];
  __shared__ __align__(16) _Float16 sW2[2 * 4 * 32 * 16];
  int tid = threadIdx.x, lane = tid & 31, wv = tid >> 5;
  stage_wT(sW0, w0, 8, 1, tid, 256);
  stage_wT(sW1, w1, HID, 2, tid, 256);
  stage_wT(sW2, w2, HID, 2, tid, 256);
  __syncthreads();
  int tile = blockIdx.x * 8 + wv;
  if (tile * 16 >= na) return;

  v16h b0;
#pragma unroll
  for (int e = 0; e < 16; ++e) b0[e] = (_Float16)0.0f;
  if (lane < 16) {  // K=8 -> elements 0..7 of lanes 0..15
    int row = tile * 16 + lane;
    if (row < na) {
      const float* ur = u + (size_t)row * 8;
#pragma unroll
      for (int e = 0; e < 8; ++e) b0[e] = (_Float16)ur[e];
    }
  }
  v8f d0[4];
  zero4(d0);
  chunk_acc(b0, 0, sW0, lane, d0);
  v8f d1[4];
  layerT64(d0, 1, sW1, lane, d1);
  v8f d2[4];
  layerT64(d1, 1, sW2, lane, d2);

  int row = tile * 16 + (lane & 15);
  int rb = (lane & 16) ? 8 : 0;
  if (row < na) {
    float* op = uenc + (size_t)row * HID + rb;
#pragma unroll
    for (int m = 0; m < 4; ++m) {
      float4 o0 = make_float4(d2[m][0], d2[m][1], d2[m][2], d2[m][3]);
      float4 o1 = make_float4(d2[m][4], d2[m][5], d2[m][6], d2[m][7]);
      *(float4*)(op + m * 16) = o0;
      *(float4*)(op + m * 16 + 4) = o1;
    }
  }
}

// ---------------- node: msg = MLP_hh(h), [64->64->64->64], bias ----------------
__global__ __launch_bounds__(256) void k_node_hh(const float* __restrict__ h, const float* w0,
                                                 const float* b0, const float* w1,
                                                 const float* b1, const float* w2,
                                                 const float* b2, float* __restrict__ msg,
                                                 int ns) {
  __shared__ __align__(16) _Float16 sW0[2 * 4 * 32 * 16];
  __shared__ __align__(16) _Float16 sW1[2 * 4 * 32 * 16];
  __shared__ __align__(16) _Float16 sW2[2 * 4 * 32 * 16];
  __shared__ __align__(16) float sb0[HID], sb1[HID], sb2[HID];
  int tid = threadIdx.x, lane = tid & 31, wv = tid >> 5;
  stage_wT(sW0, w0, HID, 2, tid, 256);
  stage_wT(sW1, w1, HID, 2, tid, 256);
  stage_wT(sW2, w2, HID, 2, tid, 256);
  stage_bias(sb0, b0, tid, 256);
  stage_bias(sb1, b1, tid, 256);
  stage_bias(sb2, b2, tid, 256);
  __syncthreads();
  int tile = blockIdx.x * 8 + wv;
  if (tile * 16 >= ns) return;

  int row = tile * 16 + (lane & 15);
  int row_eff = (row < ns) ? row : (ns - 1);
  const float* rowp = h + (size_t)row_eff * HID;
  v8f d0[4];
  zero4(d0);
#pragma unroll
  for (int kc = 0; kc < 2; ++kc) chunk_acc(frag_ld_rowf32(rowp, kc * 32, lane), kc, sW0, lane, d0);
  add_bias4(d0, sb0, lane);
  v8f d1[4];
  layerT64(d0, 1, sW1, lane, d1);
  add_bias4(d1, sb1, lane);
  v8f d2[4];
  layerT64(d1, 1, sW2, lane, d2);

  int rb = (lane & 16) ? 8 : 0;
  if (row < ns) {
    float* op = msg + (size_t)row * HID + rb;
#pragma unroll
    for (int m = 0; m < 4; ++m) {
      float4 bb0 = *(const float4*)(sb2 + m * 16 + rb);
      float4 bb1 = *(const float4*)(sb2 + m * 16 + rb + 4);
      float4 o0 = make_float4(d2[m][0] + bb0.x, d2[m][1] + bb0.y, d2[m][2] + bb0.z,
                              d2[m][3] + bb0.w);
      float4 o1 = make_float4(d2[m][4] + bb1.x, d2[m][5] + bb1.y, d2[m][6] + bb1.z,
                              d2[m][7] + bb1.w);
      *(float4*)(op + m * 16) = o0;
      *(float4*)(op + m * 16 + 4) = o1;
    }
  }
}

// Shared edge-MLP head ([srcpos(2), dstpos(2), dis] -> d2, pre-bias final).
__device__ __forceinline__ void edge_mlpT(const float* srcpos, const float* dstpos,
                                          const float* dis, const int* esrc, const int* edst,
                                          int e16, int ne, const _Float16* sW0,
                                          const float* sb0, const _Float16* sW1,
                                          const float* sb1, const _Float16* sW2, int* sS,
                                          int* sD, int lane, v8f d2[4]) {
  v16h b0;
#pragma unroll
  for (int e = 0; e < 16; ++e) b0[e] = (_Float16)0.0f;
  if (lane < 16) {
    int eid = e16 + lane;
    if (eid < ne) {
      int s = esrc[eid], t = edst[eid];
      sS[lane] = s;
      sD[lane] = t;
      b0[0] = (_Float16)srcpos[2 * s];
      b0[1] = (_Float16)srcpos[2 * s + 1];
      b0[2] = (_Float16)dstpos[2 * t];
      b0[3] = (_Float16)dstpos[2 * t + 1];
      b0[4] = (_Float16)dis[eid];
    } else {
      sS[lane] = 0;
      sD[lane] = -1;
    }
  }
  v8f d0[4];
  zero4(d0);
  chunk_acc(b0, 0, sW0, lane, d0);
  add_bias4(d0, sb0, lane);
  v8f d1[4];
  layerT64(d0, 1, sW1, lane, d1);
  add_bias4(d1, sb1, lane);
  layerT64(d1, 1, sW2, lane, d2);
}

#define EDGE_SHARED                                                                             \
  __shared__ __align__(16) _Float16 sW0[1 * 4 * 32 * 16];                                       \
  __shared__ __align__(16) _Float16 sW1[2 * 4 * 32 * 16];                                       \
  __shared__ __align__(16) _Float16 sW2[2 * 4 * 32 * 16];                                       \
  __shared__ __align__(16) float sb0[HID], sb1[HID], sb2[HID];                                  \
  __shared__ int sS[8][16], sD[8][16];

#define EDGE_STAGE(W0K)                                                                         \
  int tid = threadIdx.x, lane = tid & 31, wv = tid >> 5;                                        \
  stage_wT(sW0, w0, (W0K), 1, tid, 256);                                                        \
  stage_wT(sW1, w1, HID, 2, tid, 256);                                                          \
  stage_wT(sW2, w2, HID, 2, tid, 256);                                                          \
  stage_bias(sb0, b0, tid, 256);                                                                \
  stage_bias(sb1, b1, tid, 256);                                                                \
  stage_bias(sb2, b2, tid, 256);                                                                \
  __syncthreads();

// ---------------- a2s edges: sum_u[dst] += sigmoid(MLP_ud) * uenc[src] ----------------
__global__ __launch_bounds__(256) void k_edge_a2s(
    const float* __restrict__ pos_s, const float* __restrict__ pos_a,
    const float* __restrict__ dis, const int* __restrict__ esrc, const int* __restrict__ edst,
    const float* w0, const float* b0, const float* w1, const float* b1, const float* w2,
    const float* b2, const float* __restrict__ uenc, float* __restrict__ sum_u, int ne) {
  EDGE_SHARED
  EDGE_STAGE(5)
  int tile = blockIdx.x * 8 + wv;
  if (tile * 16 >= ne) return;

  v8f d2[4];
  edge_mlpT(pos_a, pos_s, dis, esrc, edst, tile * 16, ne, sW0, sb0, sW1, sb1, sW2, sS[wv],
            sD[wv], lane, d2);
  int row = lane & 15, rb = (lane & 16) ? 8 : 0;
  int s = sS[wv][row], t = sD[wv][row];
  if (t >= 0) {
    const float* up = uenc + (size_t)s * HID + rb;
    float* ap = sum_u + (size_t)t * HID + rb;
#pragma unroll
    for (int m = 0; m < 4; ++m) {
      float4 bb0 = *(const float4*)(sb2 + m * 16 + rb);
      float4 bb1 = *(const float4*)(sb2 + m * 16 + rb + 4);
      float4 u0 = *(const float4*)(up + m * 16);
      float4 u1 = *(const float4*)(up + m * 16 + 4);
      float bv[8] = {bb0.x, bb0.y, bb0.z, bb0.w, bb1.x, bb1.y, bb1.z, bb1.w};
      float uv[8] = {u0.x, u0.y, u0.z, u0.w, u1.x, u1.y, u1.z, u1.w};
#pragma unroll
      for (int i = 0; i < 8; ++i)
        atomicAdd(ap + m * 16 + i, fast_sigmoid(d2[m][i] + bv[i]) * uv[i]);
    }
  }
}

// ---------------- s2s pass 1: mxEnc[dst] = max(logit) ----------------
__global__ __launch_bounds__(256) void k_edge_s2s_p1(
    const float* __restrict__ pos_s, const float* __restrict__ dis,
    const int* __restrict__ esrc, const int* __restrict__ edst, const float* w0,
    const float* b0, const float* w1, const float* b1, const float* w2, const float* b2,
    unsigned* __restrict__ mxEnc, int ne) {
  EDGE_SHARED
  EDGE_STAGE(5)
  int tile = blockIdx.x * 8 + wv;
  if (tile * 16 >= ne) return;

  v8f d2[4];
  edge_mlpT(pos_s, pos_s, dis, esrc, edst, tile * 16, ne, sW0, sb0, sW1, sb1, sW2, sS[wv],
            sD[wv], lane, d2);
  int row = lane & 15, rb = (lane & 16) ? 8 : 0;
  int t = sD[wv][row];
  if (t >= 0) {
    unsigned* mp = mxEnc + (size_t)t * HID + rb;
#pragma unroll
    for (int m = 0; m < 4; ++m) {
      float4 bb0 = *(const float4*)(sb2 + m * 16 + rb);
      float4 bb1 = *(const float4*)(sb2 + m * 16 + rb + 4);
      float bv[8] = {bb0.x, bb0.y, bb0.z, bb0.w, bb1.x, bb1.y, bb1.z, bb1.w};
#pragma unroll
      for (int i = 0; i < 8; ++i) atomicMax(mp + m * 16 + i, fenc(d2[m][i] + bv[i]));
    }
  }
}

// ---------------- s2s pass 2 (recompute): z += e, acc += e*msg[src] ----------------
__global__ __launch_bounds__(256) void k_edge_s2s_p2(
    const float* __restrict__ pos_s, const float* __restrict__ dis,
    const int* __restrict__ esrc, const int* __restrict__ edst, const float* w0,
    const float* b0, const float* w1, const float* b1, const float* w2, const float* b2,
    const float* __restrict__ msg, const unsigned* __restrict__ mxEnc, float* __restrict__ zb,
    float* __restrict__ ab, int ne) {
  EDGE_SHARED
  EDGE_STAGE(5)
  int tile = blockIdx.x * 8 + wv;
  if (tile * 16 >= ne) return;

  v8f d2[4];
  edge_mlpT(pos_s, pos_s, dis, esrc, edst, tile * 16, ne, sW0, sb0, sW1, sb1, sW2, sS[wv],
            sD[wv], lane, d2);
  int row = lane & 15, rb = (lane & 16) ? 8 : 0;
  int s = sS[wv][row], t = sD[wv][row];
  if (t >= 0) {
    const unsigned* mp = mxEnc + (size_t)t * HID + rb;
    const float* gp = msg + (size_t)s * HID + rb;
    float* zp = zb + (size_t)t * HID + rb;
    float* cp = ab + (size_t)t * HID + rb;
#pragma unroll
    for (int m = 0; m < 4; ++m) {
      float4 bb0 = *(const float4*)(sb2 + m * 16 + rb);
      float4 bb1 = *(const float4*)(sb2 + m * 16 + rb + 4);
      uint4 x0 = *(const uint4*)(mp + m * 16);
      uint4 x1 = *(const uint4*)(mp + m * 16 + 4);
      float4 g0 = *(const float4*)(gp + m * 16);
      float4 g1 = *(const float4*)(gp + m * 16 + 4);
      float bv[8] = {bb0.x, bb0.y, bb0.z, bb0.w, bb1.x, bb1.y, bb1.z, bb1.w};
      unsigned xv[8] = {x0.x, x0.y, x0.z, x0.w, x1.x, x1.y, x1.z, x1.w};
      float gv[8] = {g0.x, g0.y, g0.z, g0.w, g1.x, g1.y, g1.z, g1.w};
#pragma unroll
      for (int i = 0; i < 8; ++i) {
        float ev = fast_exp(d2[m][i] + bv[i] - fdec(xv[i]));
        atomicAdd(zp + m * 16 + i, ev);
        atomicAdd(cp + m * 16 + i, ev * gv[i]);
      }
    }
  }
}

// ---------------- node update: out = MLP_up([pos, h, sum_u, acc/z]) ----------------
__device__ __forceinline__ float up_in(int node, int c, const float* ps, const float* h,
                                       const float* su, const float* zbuf, const float* abuf) {
  if (c < 2) return ps[(size_t)node * 2 + c];
  if (c < 66) return h[(size_t)node * HID + c - 2];
  if (c < 130) return su[(size_t)node * HID + c - 66];
  if (c < 194) {
    float zz = zbuf[(size_t)node * HID + c - 130];
    return (zz > 0.0f) ? abuf[(size_t)node * HID + c - 130] / zz : 0.0f;
  }
  return 0.0f;
}

__global__ __launch_bounds__(64) void k_node_up(
    const float* __restrict__ pos_s, const float* __restrict__ h,
    const float* __restrict__ sum_u, const float* __restrict__ zb, const float* __restrict__ ab,
    const float* w0, const float* b0, const float* w1, const float* b1, const float* w2,
    const float* b2, float* __restrict__ out, int ns) {
  __shared__ __align__(16) _Float16 sW0[7 * 4 * 32 * 16];  // 224x64 W^T fragments
  __shared__ __align__(16) _Float16 sW1[2 * 4 * 32 * 16];
  __shared__ __align__(16) _Float16 sW2[2 * 4 * 32 * 16];
  __shared__ __align__(16) float sb0[HID], sb1[HID], sb2[HID];
  __shared__ __align__(16) _Float16 sin[2][16 * 224];
  int tid = threadIdx.x, lane = tid & 31, wv = tid >> 5;
  stage_wT(sW0, w0, 194, 7, tid, 64);
  stage_wT(sW1, w1, HID, 2, tid, 64);
  stage_wT(sW2, w2, HID, 2, tid, 64);
  stage_bias(sb0, b0, tid, 64);
  stage_bias(sb1, b1, tid, 64);
  stage_bias(sb2, b2, tid, 64);
  __syncthreads();
  int tile = blockIdx.x * 2 + wv;
  if (tile * 16 >= ns) return;

  _Float16* buf = sin[wv];
  for (int p = lane; p < 16 * 112; p += 32) {  // paired f16 stores
    int r = p / 112, c = (p - r * 112) * 2;
    int node = tile * 16 + r;
    v2h pv;
    pv[0] = (_Float16)((node < ns) ? up_in(node, c, pos_s, h, sum_u, zb, ab) : 0.0f);
    pv[1] = (_Float16)((node < ns) ? up_in(node, c + 1, pos_s, h, sum_u, zb, ab) : 0.0f);
    *(v2h*)(buf + r * 224 + c) = pv;
  }

  const _Float16* rp = buf + (lane & 15) * 224;
  v8f d0[4];
  zero4(d0);
  for (int kc = 0; kc < 7; ++kc) {
    int kb = kc * 32 + ((lane & 16) ? 8 : 0);
    chunk_acc(frag_ld_gap(rp + kb), kc, sW0, lane, d0);
  }
  add_bias4(d0, sb0, lane);
  v8f d1[4];
  layerT64(d0, 1, sW1, lane, d1);
  add_bias4(d1, sb1, lane);
  v8f d2[4];
  layerT64(d1, 1, sW2, lane, d2);

  int row = tile * 16 + (lane & 15);
  int rb = (lane & 16) ? 8 : 0;
  if (row < ns) {
    float* op = out + (size_t)row * HID + rb;
#pragma unroll
    for (int m = 0; m < 4; ++m) {
      float4 bb0 = *(const float4*)(sb2 + m * 16 + rb);
      float4 bb1 = *(const float4*)(sb2 + m * 16 + rb + 4);
      float4 o0 = make_float4(d2[m][0] + bb0.x, d2[m][1] + bb0.y, d2[m][2] + bb0.z,
                              d2[m][3] + bb0.w);
      float4 o1 = make_float4(d2[m][4] + bb1.x, d2[m][5] + bb1.y, d2[m][6] + bb1.z,
                              d2[m][7] + bb1.w);
      *(float4*)(op + m * 16) = o0;
      *(float4*)(op + m * 16 + 4) = o1;
    }
  }
}

extern "C" void kernel_launch(void* const* d_in, const int* in_sizes, int n_in, void* d_out,
                              int out_size, void* d_ws, size_t ws_size, hipStream_t stream) {
  (void)n_in;
  (void)out_size;
  (void)ws_size;
  const float* pos_s = (const float*)d_in[0];
  const float* pos_a = (const float*)d_in[1];
  const float* h     = (const float*)d_in[2];
  const float* u     = (const float*)d_in[3];
  const int* a_src   = (const int*)d_in[4];
  const int* a_dst   = (const int*)d_in[5];
  const float* a_dis = (const float*)d_in[6];
  const int* s_src   = (const int*)d_in[7];
  const int* s_dst   = (const int*)d_in[8];
  const float* s_dis = (const float*)d_in[9];
  // params dict insertion order: ud(w,b x3), uu(w x3; None biases skipped),
  // hd(w,b x3), hh(w,b x3), up(w,b x3)
  const float* ud_w0 = (const float*)d_in[10];
  const float* ud_b0 = (const float*)d_in[11];
  const float* ud_w1 = (const float*)d_in[12];
  const float* ud_b1 = (const float*)d_in[13];
  const float* ud_w2 = (const float*)d_in[14];
  const float* ud_b2 = (const float*)d_in[15];
  const float* uu_w0 = (const float*)d_in[16];
  const float* uu_w1 = (const float*)d_in[17];
  const float* uu_w2 = (const float*)d_in[18];
  const float* hd_w0 = (const float*)d_in[19];
  const float* hd_b0 = (const float*)d_in[20];
  const float* hd_w1 = (const float*)d_in[21];
  const float* hd_b1 = (const float*)d_in[22];
  const float* hd_w2 = (const float*)d_in[23];
  const float* hd_b2 = (const float*)d_in[24];
  const float* hh_w0 = (const float*)d_in[25];
  const float* hh_b0 = (const float*)d_in[26];
  const float* hh_w1 = (const float*)d_in[27];
  const float* hh_b1 = (const float*)d_in[28];
  const float* hh_w2 = (const float*)d_in[29];
  const float* hh_b2 = (const float*)d_in[30];
  const float* up_w0 = (const float*)d_in[31];
  const float* up_b0 = (const float*)d_in[32];
  const float* up_w1 = (const float*)d_in[33];
  const float* up_b1 = (const float*)d_in[34];
  const float* up_w2 = (const float*)d_in[35];
  const float* up_b2 = (const float*)d_in[36];

  int ns = in_sizes[0] / 2, na = in_sizes[1] / 2;
  int ea = in_sizes[4], es = in_sizes[7];

  float* uenc   = (float*)d_ws;                            // [na,64]
  float* msg    = uenc + (size_t)na * HID;                 // [ns,64]
  float* sum_u  = msg + (size_t)ns * HID;                  // [ns,64]  zeroed
  unsigned* mxE = (unsigned*)(sum_u + (size_t)ns * HID);   // [ns,64]  zeroed (0 == -inf enc)
  float* zb     = (float*)(mxE + (size_t)ns * HID);        // [ns,64]  zeroed
  float* ab     = zb + (size_t)ns * HID;                   // [ns,64]  zeroed
  hipMemsetAsync(sum_u, 0, (size_t)ns * HID * sizeof(float) * 4, stream);

  int t_na = (na + 15) / 16, t_ns = (ns + 15) / 16;
  int t_ea = (ea + 15) / 16, t_es = (es + 15) / 16;

  k_node_uu<<<(t_na + 7) / 8, 256, 0, stream>>>(u, uu_w0, uu_w1, uu_w2, uenc, na);
  k_node_hh<<<(t_ns + 7) / 8, 256, 0, stream>>>(h, hh_w0, hh_b0, hh_w1, hh_b1, hh_w2, hh_b2,
                                                msg, ns);
  k_edge_a2s<<<(t_ea + 7) / 8, 256, 0, stream>>>(pos_s, pos_a, a_dis, a_src, a_dst, ud_w0,
                                                 ud_b0, ud_w1, ud_b1, ud_w2, ud_b2, uenc,
                                                 sum_u, ea);
  k_edge_s2s_p1<<<(t_es + 7) / 8, 256, 0, stream>>>(pos_s, s_dis, s_src, s_dst, hd_w0, hd_b0,
                                                    hd_w1, hd_b1, hd_w2, hd_b2, mxE, es);
  k_edge_s2s_p2<<<(t_es + 7) / 8, 256, 0, stream>>>(pos_s, s_dis, s_src, s_dst, hd_w0, hd_b0,
                                                    hd_w1, hd_b1, hd_w2, hd_b2, msg, mxE, zb,
                                                    ab, es);
  k_node_up<<<(t_ns + 1) / 2, 64, 0, stream>>>(pos_s, h, sum_u, zb, ab, up_w0, up_b0, up_w1,
                                               up_b1, up_w2, up_b2, (float*)d_out, ns);
}
